// GraphEncoder_81011673137443
// MI455X (gfx1250) — compile-verified, matrix operand order checked by reading, not run on hardware
//
#include <hip/hip_runtime.h>
#include <hip/hip_bf16.h>

typedef __attribute__((ext_vector_type(16))) _Float16 v16h;
typedef __attribute__((ext_vector_type(8)))  _Float16 v8h;
typedef __attribute__((ext_vector_type(8)))  float    v8f;

#define ND 256  // hidden dim D (== K for all GEMMs)

// ---------------------------------------------------------------------------
// WMMA fragment loaders (16-bit A-matrix 16x32 layout, wave32):
//   lanes 0-15 : row = lane,    halves = K[0..7]   and K[16..23]
//   lanes 16-31: row = lane-16, halves = K[8..15]  and K[24..31]
// Weights are pre-transposed to [N][K] f16 so the B fragment uses the same
// loader with row = n (B^T stored row-major).
// ---------------------------------------------------------------------------
__device__ __forceinline__ v16h load_frag_w(const _Float16* __restrict__ P,
                                            int r0, int k0) {
  const int lane = threadIdx.x & 31;
  const int r  = r0 + (lane & 15);
  const int kk = k0 + ((lane & 16) ? 8 : 0);
  const _Float16* q = P + ((size_t)r << 8) + kk;   // ld = 256 halves
  const v8h lo = *(const v8h*)(q);
  const v8h hi = *(const v8h*)(q + 16);
  v16h f;
#pragma unroll
  for (int i = 0; i < 8; ++i) { f[i] = lo[i]; f[8 + i] = hi[i]; }
  return f;
}

__device__ __forceinline__ v16h load_frag_a(const _Float16* __restrict__ P,
                                            int r0, int k0, int M) {
  const int lane = threadIdx.x & 31;
  int r = r0 + (lane & 15);
  r = (r < M) ? r : (M - 1);                        // clamp tail rows
  const int kk = k0 + ((lane & 16) ? 8 : 0);
  const _Float16* q = P + ((size_t)r << 8) + kk;
  const v8h lo = *(const v8h*)(q);
  const v8h hi = *(const v8h*)(q + 16);
  v16h f;
#pragma unroll
  for (int i = 0; i < 8; ++i) { f[i] = lo[i]; f[8 + i] = hi[i]; }
  return f;
}

template<int ACT>
__device__ __forceinline__ float act_apply(float v) {
  if constexpr (ACT == 1) return fmaxf(v, 0.f);                                 // ReLU
  else if constexpr (ACT == 2) return 0.5f * v * (1.f + erff(v * 0.70710678118654752f)); // exact GELU
  else return v;
}

// ---------------------------------------------------------------------------
// Y[M,256] = act(A[M,256] @ W[256,256] + bias), W given transposed f16 [N][K].
// Block: 256 threads = 8 waves; block tile 128 rows x 64 cols (grid.y = 4).
// Wave w: rows r0..r0+15, 4 col tiles -> 4x v8f accumulators.
// K loop is software-pipelined: next-step A + 4xB fragments are fetched into
// distinct registers while the current step's 4 WMMAs issue, so loads overlap
// matrix ops instead of draining loadcnt before every v_wmma.
// ---------------------------------------------------------------------------
template<int ACT, bool WF32, bool WF16>
__global__ __launch_bounds__(256) void gemm_d256(
    const _Float16* __restrict__ A, const _Float16* __restrict__ Wt,
    const float* __restrict__ bias, float* __restrict__ Of,
    _Float16* __restrict__ Oh, int M)
{
  const int lane = threadIdx.x & 31;
  const int wave = threadIdx.x >> 5;
  const int r0 = blockIdx.x * 128 + wave * 16;
  const int c0 = blockIdx.y * 64;

  v8f acc[4] = {v8f{}, v8f{}, v8f{}, v8f{}};

  // prologue: fragments for k0 = 0
  v16h a_cur = load_frag_a(A, r0, 0, M);
  v16h b_cur[4];
#pragma unroll
  for (int c = 0; c < 4; ++c) b_cur[c] = load_frag_w(Wt, c0 + c * 16, 0);

#pragma unroll
  for (int k0 = 0; k0 < ND; k0 += 32) {
    v16h a_nxt{};
    v16h b_nxt[4] = {v16h{}, v16h{}, v16h{}, v16h{}};
    if (k0 + 32 < ND) {                       // prefetch next K-step
      a_nxt = load_frag_a(A, r0, k0 + 32, M);
#pragma unroll
      for (int c = 0; c < 4; ++c) b_nxt[c] = load_frag_w(Wt, c0 + c * 16, k0 + 32);
    }
#pragma unroll
    for (int c = 0; c < 4; ++c) {
      acc[c] = __builtin_amdgcn_wmma_f32_16x16x32_f16(
          false, a_cur, false, b_cur[c], (short)0, acc[c], false, false);
    }
    a_cur = a_nxt;
#pragma unroll
    for (int c = 0; c < 4; ++c) b_cur[c] = b_nxt[c];
  }

  // C/D layout: VGPR i, lanes 0-15: row i, col lane; lanes 16-31: row i+8.
  const int rb = r0 + ((lane >> 4) << 3);
#pragma unroll
  for (int c = 0; c < 4; ++c) {
    const int col = c0 + c * 16 + (lane & 15);
    const float bv = bias[col];
#pragma unroll
    for (int i = 0; i < 8; ++i) {
      const int row = rb + i;
      if (row < M) {
        const float v = act_apply<ACT>(acc[c][i] + bv);
        const size_t o = ((size_t)row << 8) + col;
        if constexpr (WF32) Of[o] = v;
        if constexpr (WF16) Oh[o] = (_Float16)v;
      }
    }
  }
}

// ---------------------------------------------------------------------------
// fp32 [K][N] weight -> f16 transposed [N][K]
// ---------------------------------------------------------------------------
__global__ void w_to_f16t(const float* __restrict__ W, _Float16* __restrict__ Wt) {
  const int k = blockIdx.x, n = threadIdx.x;
  Wt[n * ND + k] = (_Float16)W[k * ND + n];
}

__device__ __forceinline__ float block_sum256(float v, float* sbuf) {
  const int t = threadIdx.x;
  sbuf[t] = v; __syncthreads();
#pragma unroll
  for (int s = 128; s > 0; s >>= 1) {
    if (t < s) sbuf[t] += sbuf[t + s];
    __syncthreads();
  }
  const float r = sbuf[0]; __syncthreads();
  return r;
}

// ---------------------------------------------------------------------------
// Per-row embedding sum (F columns) + LayerNorm -> f16. One block per row.
// ---------------------------------------------------------------------------
__global__ __launch_bounds__(256) void encode_embed_ln(
    const int* __restrict__ idx, int F, int V,
    const float* __restrict__ emb, const float* __restrict__ g,
    const float* __restrict__ b, _Float16* __restrict__ out16)
{
  __shared__ float sbuf[256];
  const int n = blockIdx.x, c = threadIdx.x;
  float s = 0.f;
  for (int f = 0; f < F; ++f) {
    const int id = idx[n * F + f];
    s += emb[((size_t)f * V + id) * ND + c];
  }
  const float mean = block_sum256(s, sbuf) * (1.f / ND);
  const float var  = block_sum256(s * s, sbuf) * (1.f / ND) - mean * mean;
  const float y = (s - mean) * rsqrtf(var + 1e-5f) * g[c] + b[c];
  out16[((size_t)n << 8) + c] = (_Float16)y;
}

// m = relu(h[src] + eL[e]); agg[dst] += m  (native f32 atomics). Block = edge.
__global__ __launch_bounds__(256) void edge_message_scatter(
    const float* __restrict__ h, const float* __restrict__ eL,
    const int* __restrict__ src, const int* __restrict__ dst,
    float* __restrict__ agg)
{
  const int e = blockIdx.x, c = threadIdx.x;
  const int s = src[e], d = dst[e];
  float m = h[((size_t)s << 8) + c] + eL[((size_t)e << 8) + c];
  m = fmaxf(m, 0.f);
  unsafeAtomicAdd(&agg[((size_t)d << 8) + c], m);
}

// o16 = f16(agg + h)   (GINE: (1+eps)*x + aggr, eps = 0)
__global__ __launch_bounds__(256) void add_to_f16(
    const float* __restrict__ a, const float* __restrict__ b,
    _Float16* __restrict__ out)
{
  const size_t i = ((size_t)blockIdx.x << 8) + threadIdx.x;
  out[i] = (_Float16)(a[i] + b[i]);
}

// h = LN(o + h_in) * g + b, in place over h. One block per row.
__global__ __launch_bounds__(256) void ln_residual(
    const float* __restrict__ o, float* __restrict__ h,
    const float* __restrict__ g, const float* __restrict__ b)
{
  __shared__ float sbuf[256];
  const int n = blockIdx.x, c = threadIdx.x;
  const size_t i = ((size_t)n << 8) + c;
  const float x = o[i] + h[i];
  const float mean = block_sum256(x, sbuf) * (1.f / ND);
  const float var  = block_sum256(x * x, sbuf) * (1.f / ND) - mean * mean;
  h[i] = (x - mean) * rsqrtf(var + 1e-5f) * g[c] + b[c];
}

__global__ __launch_bounds__(256) void pool_scatter(
    const float* __restrict__ h, const int* __restrict__ batch,
    float* __restrict__ sums, float* __restrict__ cnt)
{
  const int n = blockIdx.x, c = threadIdx.x;
  const int gidx = batch[n];
  unsafeAtomicAdd(&sums[((size_t)gidx << 8) + c], h[((size_t)n << 8) + c]);
  if (c == 0) unsafeAtomicAdd(&cnt[gidx], 1.f);
}

// mean -> @proj_W + b -> L2-normalize. One block per graph (tiny GEMM, VALU).
__global__ __launch_bounds__(256) void proj_norm(
    const float* __restrict__ sums, const float* __restrict__ cnt,
    const float* __restrict__ W, const float* __restrict__ b,
    float* __restrict__ out)
{
  __shared__ float mean[256];
  __shared__ float sbuf[256];
  const int g = blockIdx.x, c = threadIdx.x;
  const float inv = 1.f / fmaxf(cnt[g], 1.f);
  mean[c] = sums[((size_t)g << 8) + c] * inv;
  __syncthreads();
  float acc = b[c];
  for (int k = 0; k < ND; ++k) acc += mean[k] * W[k * ND + c];
  const float ss = block_sum256(acc * acc, sbuf);
  const float nrm = fmaxf(sqrtf(ss), 1e-12f);
  out[((size_t)g << 8) + c] = acc / nrm;
}

// ---------------------------------------------------------------------------
extern "C" void kernel_launch(void* const* d_in, const int* in_sizes, int n_in,
                              void* d_out, int out_size, void* d_ws, size_t ws_size,
                              hipStream_t stream) {
  (void)in_sizes; (void)n_in; (void)out_size; (void)ws_size;
  const int N = 100000, E = 320000, G = 4096, L = 4;
  const int FA = 9, VA = 128, FB = 3, VB = 16;

  const int*   x         = (const int*)d_in[0];
  const int*   edge_attr = (const int*)d_in[1];
  const int*   src       = (const int*)d_in[2];
  const int*   dst       = src + E;
  const int*   batch     = (const int*)d_in[3];
  const float* atom_emb  = (const float*)d_in[4];
  const float* atom_ln_g = (const float*)d_in[5];
  const float* atom_ln_b = (const float*)d_in[6];
  const float* atom_W1   = (const float*)d_in[7];
  const float* atom_b1   = (const float*)d_in[8];
  const float* atom_W2   = (const float*)d_in[9];
  const float* atom_b2   = (const float*)d_in[10];
  const float* bond_emb  = (const float*)d_in[11];
  const float* bond_ln_g = (const float*)d_in[12];
  const float* bond_ln_b = (const float*)d_in[13];
  const float* bond_W1   = (const float*)d_in[14];
  const float* bond_b1   = (const float*)d_in[15];
  const float* bond_W2   = (const float*)d_in[16];
  const float* bond_b2   = (const float*)d_in[17];
  const float* conv_linW = (const float*)d_in[18];
  const float* conv_linb = (const float*)d_in[19];
  const float* conv_W1   = (const float*)d_in[20];
  const float* conv_b1   = (const float*)d_in[21];
  const float* conv_W2   = (const float*)d_in[22];
  const float* conv_b2   = (const float*)d_in[23];
  const float* ln_g      = (const float*)d_in[24];
  const float* ln_b      = (const float*)d_in[25];
  const float* proj_W    = (const float*)d_in[26];
  const float* proj_b    = (const float*)d_in[27];

  // ---- workspace carve (256B aligned) ----
  char* ws = (char*)d_ws;
  size_t off = 0;
  auto carve = [&](size_t bytes) -> char* {
    char* p = ws + off;
    off = (off + bytes + 255) & ~(size_t)255;
    return p;
  };
  const size_t MAT = (size_t)ND * ND;                     // 65536 elems
  _Float16* wt   = (_Float16*)carve(16 * MAT * sizeof(_Float16)); // 16 f16 W^T
  float*    h    = (float*)   carve((size_t)N * ND * 4);
  float*    agg  = (float*)   carve((size_t)N * ND * 4);  // also MLP2 output
  _Float16* o16  = (_Float16*)carve((size_t)N * ND * 2);
  _Float16* t16  = (_Float16*)carve((size_t)N * ND * 2);
  _Float16* e16  = (_Float16*)carve((size_t)E * ND * 2);
  float*    eL   = (float*)   carve((size_t)E * ND * 4);  // per-layer edge lin
  float*    sums = (float*)   carve((size_t)G * ND * 4);
  float*    cnt  = (float*)   carve((size_t)G * 4);
  // bond-encoder f16 staging aliases the eL region (used before layers begin)
  _Float16* eln16 = (_Float16*)eL;
  _Float16* et16  = eln16 + (size_t)E * ND;

  // ---- weight prep: fp32 [K][N] -> f16 [N][K] ----
  // slots: 0 atomW1, 1 atomW2, 2 bondW1, 3 bondW2, 4+l linW, 8+l W1, 12+l W2
  w_to_f16t<<<ND, ND, 0, stream>>>(atom_W1, wt + 0 * MAT);
  w_to_f16t<<<ND, ND, 0, stream>>>(atom_W2, wt + 1 * MAT);
  w_to_f16t<<<ND, ND, 0, stream>>>(bond_W1, wt + 2 * MAT);
  w_to_f16t<<<ND, ND, 0, stream>>>(bond_W2, wt + 3 * MAT);
  for (int l = 0; l < L; ++l) {
    w_to_f16t<<<ND, ND, 0, stream>>>(conv_linW + l * MAT, wt + (4 + l) * MAT);
    w_to_f16t<<<ND, ND, 0, stream>>>(conv_W1   + l * MAT, wt + (8 + l) * MAT);
    w_to_f16t<<<ND, ND, 0, stream>>>(conv_W2   + l * MAT, wt + (12 + l) * MAT);
  }

  const dim3 blk(256);
  const dim3 gN((N + 127) / 128, 4);
  const dim3 gE((E + 127) / 128, 4);

  // ---- atom encoder: embed+LN -> GELU MLP -> h (f32) ----
  encode_embed_ln<<<N, blk, 0, stream>>>(x, FA, VA, atom_emb, atom_ln_g, atom_ln_b, o16);
  gemm_d256<2, false, true><<<gN, blk, 0, stream>>>(o16, wt + 0 * MAT, atom_b1, nullptr, t16, N);
  gemm_d256<0, true, false><<<gN, blk, 0, stream>>>(t16, wt + 1 * MAT, atom_b2, h, nullptr, N);

  // ---- bond encoder: embed+LN -> GELU MLP -> e16 (f16, reused all layers) ----
  encode_embed_ln<<<E, blk, 0, stream>>>(edge_attr, FB, VB, bond_emb, bond_ln_g, bond_ln_b, eln16);
  gemm_d256<2, false, true><<<gE, blk, 0, stream>>>(eln16, wt + 2 * MAT, bond_b1, nullptr, et16, E);
  gemm_d256<0, false, true><<<gE, blk, 0, stream>>>(et16, wt + 3 * MAT, bond_b2, nullptr, e16, E);

  // ---- GINE layers ----
  for (int l = 0; l < L; ++l) {
    gemm_d256<0, true, false><<<gE, blk, 0, stream>>>(
        e16, wt + (4 + l) * MAT, conv_linb + l * ND, eL, nullptr, E);
    hipMemsetAsync(agg, 0, (size_t)N * ND * 4, stream);
    edge_message_scatter<<<E, blk, 0, stream>>>(h, eL, src, dst, agg);
    add_to_f16<<<N, blk, 0, stream>>>(agg, h, o16);           // o16 = f16(agg + h)
    gemm_d256<1, false, true><<<gN, blk, 0, stream>>>(
        o16, wt + (8 + l) * MAT, conv_b1 + l * ND, nullptr, t16, N);
    gemm_d256<2, true, false><<<gN, blk, 0, stream>>>(        // gelu fused
        t16, wt + (12 + l) * MAT, conv_b2 + l * ND, agg, nullptr, N);
    ln_residual<<<N, blk, 0, stream>>>(agg, h, ln_g + l * ND, ln_b + l * ND);
  }

  // ---- global mean pool + projection + L2 normalize ----
  hipMemsetAsync(sums, 0, (size_t)G * ND * 4, stream);
  hipMemsetAsync(cnt, 0, (size_t)G * 4, stream);
  pool_scatter<<<N, blk, 0, stream>>>(h, batch, sums, cnt);
  proj_norm<<<G, blk, 0, stream>>>(sums, cnt, proj_W, proj_b, (float*)d_out);
}
